// Generator_5299989643721
// MI455X (gfx1250) — matrix operand order repaired and checked by reading, not compiled, not run on hardware
//
#include <hip/hip_runtime.h>
#include <math.h>
#include <stdint.h>

typedef __attribute__((ext_vector_type(16))) _Float16 v16h;
typedef __attribute__((ext_vector_type(8)))  float    v8f;

#define SQRT2F 1.41421356237f
#define INV_SQRT512 0.044194173824159216f

union FragH { v16h v; uint4 q[2]; };
union FragF { v8f  v; float f[8]; };

struct MlpPtrs { const float* w[8]; const float* b[8]; };

// ---------------- Style MLP: PixelNorm + 8x EqualLinear(lr_mul=0.01, fused lrelu) ----
__global__ __launch_bounds__(512)
void style_mlp_kernel(const float* __restrict__ z, MlpPtrs P, float* __restrict__ wlat)
{
    __shared__ float xs[512];
    __shared__ float red[16];
    const int b = blockIdx.x;
    const int t = threadIdx.x;
    const int lane = t & 31, wid = t >> 5;

    float v = z[b * 512 + t];
    float sq = v * v;
    #pragma unroll
    for (int off = 16; off > 0; off >>= 1) sq += __shfl_down(sq, off, 32);
    if (lane == 0) red[wid] = sq;
    __syncthreads();
    if (t == 0) { float tot = 0.f; for (int i = 0; i < 16; i++) tot += red[i]; red[0] = tot; }
    __syncthreads();
    v *= rsqrtf(red[0] * (1.f / 512.f) + 1e-8f);
    xs[t] = v;
    __syncthreads();

    const float scale = 0.01f * INV_SQRT512;
    for (int l = 0; l < 8; l++) {
        const float* W = P.w[l] + (size_t)t * 512;
        float acc = 0.f;
        for (int j = 0; j < 512; j++) acc += xs[j] * W[j];
        acc = acc * scale + P.b[l][t] * 0.01f;
        acc = (acc < 0.f ? 0.2f * acc : acc) * SQRT2F;
        __syncthreads();
        xs[t] = acc;
        __syncthreads();
    }
    wlat[b * 512 + t] = xs[t];
}

// ---------------- s = EqualLinear(wlat; mod_w, mod_b)  [B, Cin] -------------------
__global__ void style_proj_kernel(const float* __restrict__ wlat,
                                  const float* __restrict__ mod_w,
                                  const float* __restrict__ mod_b,
                                  float* __restrict__ s, int Cin)
{
    int ci = blockIdx.x * blockDim.x + threadIdx.x;
    int b  = blockIdx.y;
    if (ci >= Cin) return;
    const float* x = wlat + b * 512;
    const float* W = mod_w + (size_t)ci * 512;
    float acc = 0.f;
    for (int j = 0; j < 512; j++) acc += x[j] * W[j];
    s[b * Cin + ci] = acc * INV_SQRT512 + mod_b[ci];
}

// ---------------- modulate + demodulate (+optional kernel flip) -> f16 A matrix ----
__global__ __launch_bounds__(256)
void modulate_kernel(const float* __restrict__ w, const float* __restrict__ s,
                     _Float16* __restrict__ A, int Cin, int Cout, int flip, float scale)
{
    const int co = blockIdx.x, b = blockIdx.y, t = threadIdx.x;
    const int K = Cin * 9;
    const float* wrow = w + (size_t)co * K;
    const float* sb   = s + b * Cin;
    __shared__ float red[8];

    float ss = 0.f;
    for (int k = t; k < K; k += 256) {
        int ci = k / 9;
        float v = scale * wrow[k] * sb[ci];
        ss += v * v;
    }
    int lane = t & 31, wid = t >> 5;
    #pragma unroll
    for (int off = 16; off > 0; off >>= 1) ss += __shfl_down(ss, off, 32);
    if (lane == 0) red[wid] = ss;
    __syncthreads();
    if (t == 0) { float tot = 0.f; for (int i = 0; i < 8; i++) tot += red[i]; red[0] = rsqrtf(tot + 1e-8f); }
    __syncthreads();
    const float d = red[0];

    _Float16* Arow = A + ((size_t)b * Cout + co) * K;
    for (int k = t; k < K; k += 256) {
        int ci = k / 9, tap = k - ci * 9;
        float v = scale * wrow[k] * sb[ci] * d;
        int ks = ci * 9 + (flip ? (8 - tap) : tap);
        Arow[ks] = (_Float16)v;
    }
}

// ---------------- implicit-GEMM modulated conv (3x3, normal or lhs-dilated x2) -----
// Out = A[b][Cout][Cin*9] x im2col(X[b][Cin][Hin][Win])
// Block tile 128(M) x 64(N), K-step 32; 4 waves x (4x2 WMMA tiles each).
// A tiles: GLOBAL_LOAD_ASYNC_TO_LDS_B128, double-buffered, overlapped with compute.
// fuse=1 (non-upsample): noise+bias+lrelu fused, f16 activations written directly.
#define ALD 40   // LDS row stride in f16 (32 + 8 pad)

__global__ __launch_bounds__(128)
void modconv_wmma_kernel(const _Float16* __restrict__ A,
                         const _Float16* __restrict__ X,
                         float* __restrict__ Out,
                         const float* __restrict__ noise,
                         const float* __restrict__ nw,
                         const float* __restrict__ ab,
                         _Float16* __restrict__ actOut,
                         int Cin, int Cout, int Hin, int Win,
                         int Hout, int Wout, int upsample, int fuse)
{
    __shared__ _Float16 As[2][128][ALD];
    __shared__ _Float16 Bs[2][64][ALD];

    const int K  = Cin * 9;
    const int Np = Hout * Wout;
    const int b      = blockIdx.z;
    const int coBase = blockIdx.y * 128;
    const int nBase  = blockIdx.x * 64;
    const int tid  = threadIdx.x;
    const int lane = tid & 31;
    const int wid  = tid >> 5;
    const int waveM = (wid >> 1) * 64;   // 0 or 64
    const int waveN = (wid & 1) * 32;    // 0 or 32

    const _Float16* Ab = A + ((size_t)b * Cout + coBase) * K;
    const _Float16* Xb = X + (size_t)b * Cin * Hin * Win;

    // fixed im2col coordinates for this thread's B-tile column
    const int nl    = tid & 63;
    const int khalf = (tid >> 6) << 4;   // 0 or 16
    const int n      = nBase + nl;
    const bool nValid = (n < Np);
    const int oy = nValid ? (n / Wout) : 0;
    const int ox = nValid ? (n - oy * Wout) : 0;

    FragF acc[4][2];
    #pragma unroll
    for (int i = 0; i < 4; i++)
        #pragma unroll
        for (int j = 0; j < 2; j++)
            #pragma unroll
            for (int k = 0; k < 8; k++) acc[i][j].f[k] = 0.f;

    const unsigned ldsA[2] = { (unsigned)(uintptr_t)(&As[0][tid][0]),
                               (unsigned)(uintptr_t)(&As[1][tid][0]) };
    const _Float16* aSrc = Ab + (size_t)tid * K;

    // prologue: async-stage A tile 0
    {
        uint64_t gaddr = (uint64_t)(const void*)(aSrc);
        asm volatile(
            "global_load_async_to_lds_b128 %0, %1, off offset:0\n"
            "global_load_async_to_lds_b128 %0, %1, off offset:16\n"
            "global_load_async_to_lds_b128 %0, %1, off offset:32\n"
            "global_load_async_to_lds_b128 %0, %1, off offset:48\n"
            :: "v"(ldsA[0]), "v"(gaddr) : "memory");
    }

    int buf = 0;
    for (int k0 = 0; k0 < K; k0 += 32) {
        // ---- gather B tile (im2col, zero-insert sampling for upsample) ----
        {
            int ks  = k0 + khalf;
            int ci  = ks / 9;
            int tap = ks - ci * 9;
            #pragma unroll 4
            for (int j = 0; j < 16; j++) {
                float v = 0.f;
                if (nValid) {
                    int ky = tap / 3, kx = tap - (tap / 3) * 3;
                    if (!upsample) {
                        int iy = oy + ky - 1, ix = ox + kx - 1;
                        if ((unsigned)iy < (unsigned)Hin && (unsigned)ix < (unsigned)Win)
                            v = (float)Xb[((size_t)ci * Hin + iy) * Win + ix];
                    } else {
                        int py = oy + ky - 2, px = ox + kx - 2;
                        if (py >= 0 && px >= 0 && !(py & 1) && !(px & 1)) {
                            int iy = py >> 1, ix = px >> 1;
                            if (iy < Hin && ix < Win)
                                v = (float)Xb[((size_t)ci * Hin + iy) * Win + ix];
                        }
                    }
                }
                Bs[buf][nl][khalf + j] = (_Float16)v;
                if (++tap == 9) { tap = 0; ++ci; }
            }
        }

        // A tile for this step is the only outstanding async group
        asm volatile("s_wait_asynccnt 0" ::: "memory");
        __syncthreads();   // all A rows + B gather visible; prev compute finished

        // issue async A copy for next K-step into the other buffer
        if (k0 + 32 < K) {
            uint64_t gaddr = (uint64_t)(const void*)(aSrc + k0 + 32);
            asm volatile(
                "global_load_async_to_lds_b128 %0, %1, off offset:0\n"
                "global_load_async_to_lds_b128 %0, %1, off offset:16\n"
                "global_load_async_to_lds_b128 %0, %1, off offset:32\n"
                "global_load_async_to_lds_b128 %0, %1, off offset:48\n"
                :: "v"(ldsA[buf ^ 1]), "v"(gaddr) : "memory");
        }

        // ---- WMMA: 4x2 tiles of 16x16 per wave ----
        const int lrow = lane & 15;
        const int cA = (lane >> 4) << 3;    // A: K chunks {c, c+16}
        const int cB = (lane >> 4) << 3;    // B: K chunks {c, c+16}

        FragH af[4], bf[2];
        #pragma unroll
        for (int mi = 0; mi < 4; mi++) {
            const _Float16* ap = &As[buf][waveM + mi * 16 + lrow][0];
            af[mi].q[0] = *(const uint4*)(ap + cA);
            af[mi].q[1] = *(const uint4*)(ap + cA + 16);
        }
        #pragma unroll
        for (int ni = 0; ni < 2; ni++) {
            const _Float16* bp = &Bs[buf][waveN + ni * 16 + lrow][0];
            bf[ni].q[0] = *(const uint4*)(bp + cB);
            bf[ni].q[1] = *(const uint4*)(bp + cB + 16);
        }
        #pragma unroll
        for (int mi = 0; mi < 4; mi++)
            #pragma unroll
            for (int ni = 0; ni < 2; ni++)
                acc[mi][ni].v = __builtin_amdgcn_wmma_f32_16x16x32_f16(
                    false, af[mi].v, false, bf[ni].v,
                    (short)0, acc[mi][ni].v, false, false);
        buf ^= 1;
    }

    // ---- epilogue (C layout: M = vgpr + 8*(lane>=16), N = lane%16) ----
    const int lrow = lane & 15;
    const int mOff = (lane >> 4) << 3;
    if (fuse) {
        _Float16* actB = actOut + (size_t)b * Cout * Np;
        const float nwv = nw[0];
        for (int mi = 0; mi < 4; mi++)
            for (int ni = 0; ni < 2; ni++) {
                int nIdx = nBase + waveN + ni * 16 + lrow;
                if (nIdx < Np) {
                    float nz = nwv * noise[nIdx];
                    int mRow = coBase + waveM + mi * 16 + mOff;
                    #pragma unroll
                    for (int vv = 0; vv < 8; vv++) {
                        float v = acc[mi][ni].f[vv] + nz + ab[mRow + vv];
                        v = (v < 0.f ? 0.2f * v : v) * SQRT2F;
                        actB[(size_t)(mRow + vv) * Np + nIdx] = (_Float16)v;
                    }
                }
            }
    } else {
        float* Outb = Out + (size_t)b * Cout * Np;
        for (int mi = 0; mi < 4; mi++)
            for (int ni = 0; ni < 2; ni++) {
                int nIdx = nBase + waveN + ni * 16 + lrow;
                if (nIdx < Np) {
                    int mRow = coBase + waveM + mi * 16 + mOff;
                    #pragma unroll
                    for (int vv = 0; vv < 8; vv++)
                        Outb[(size_t)(mRow + vv) * Np + nIdx] = acc[mi][ni].f[vv];
                }
            }
    }
}

// ---------------- blur(4x4, pad(1,1)) + noise + bias + lrelu  ((2H+1)^2 -> (2H)^2) -
__global__ void blur_act_kernel(const float* __restrict__ g, const float* __restrict__ noise,
                                const float* __restrict__ nw, const float* __restrict__ ab,
                                _Float16* __restrict__ act, int C, int H, int S, int total)
{
    int i = blockIdx.x * 256 + threadIdx.x;
    if (i >= total) return;
    int x = i % H;
    int y = (i / H) % H;
    int c = (i / (H * H)) % C;
    int b = i / (H * H * C);
    const float* src = g + (size_t)(b * C + c) * S * S;
    const float cw[4] = {0.25f, 0.75f, 0.75f, 0.25f};
    float accv = 0.f;
    #pragma unroll
    for (int ty = 0; ty < 4; ty++) {
        int sy = y + ty - 1;
        if ((unsigned)sy >= (unsigned)S) continue;
        float rowacc = 0.f;
        #pragma unroll
        for (int tx = 0; tx < 4; tx++) {
            int sx = x + tx - 1;
            if ((unsigned)sx >= (unsigned)S) continue;
            rowacc += cw[tx] * src[sy * S + sx];
        }
        accv += cw[ty] * rowacc;
    }
    accv += nw[0] * noise[y * H + x] + ab[c];
    accv = (accv < 0.f ? 0.2f * accv : accv) * SQRT2F;
    act[i] = (_Float16)accv;
}

// ---------------- to_rgb: modulated 1x1 (no demod) + bias + upsampled skip ---------
__global__ __launch_bounds__(256)
void to_rgb_kernel(const _Float16* __restrict__ act, const float* __restrict__ w,
                   const float* __restrict__ s, const float* __restrict__ bias,
                   const float* __restrict__ skip, float* __restrict__ out,
                   int Cin, int H, int Hs, float scale)
{
    __shared__ float weff[3 * 512];
    const int b = blockIdx.y;
    const int t = threadIdx.x;
    for (int idx = t; idx < 3 * Cin; idx += 256) {
        int ci = idx % Cin;
        weff[idx] = w[idx] * scale * s[b * Cin + ci];
    }
    __syncthreads();

    const int HW = H * H;
    int p = blockIdx.x * 256 + t;
    if (p >= HW) return;
    int y = p / H, x = p - y * H;

    float r3[3] = {bias[0], bias[1], bias[2]};
    const _Float16* ap = act + (size_t)b * Cin * HW;
    for (int ci = 0; ci < Cin; ci++) {
        float av = (float)ap[(size_t)ci * HW + p];
        r3[0] += av * weff[ci];
        r3[1] += av * weff[Cin + ci];
        r3[2] += av * weff[2 * Cin + ci];
    }

    if (skip) {
        // upfirdn2d(skip, 4*BLUR_K, up=2, pad=(2,1)): 2 taps per axis per parity
        int qy[2]; float wy[2];
        int qx[2]; float wx[2];
        if (y & 1) { qy[0] = (y - 1) >> 1; wy[0] = 0.75f; qy[1] = (y + 1) >> 1; wy[1] = 0.25f; }
        else       { qy[0] = (y - 2) >> 1; wy[0] = 0.25f; qy[1] = y >> 1;       wy[1] = 0.75f; }
        if (x & 1) { qx[0] = (x - 1) >> 1; wx[0] = 0.75f; qx[1] = (x + 1) >> 1; wx[1] = 0.25f; }
        else       { qx[0] = (x - 2) >> 1; wx[0] = 0.25f; qx[1] = x >> 1;       wx[1] = 0.75f; }
        for (int c = 0; c < 3; c++) {
            const float* sp = skip + (size_t)(b * 3 + c) * Hs * Hs;
            float accs = 0.f;
            #pragma unroll
            for (int a2 = 0; a2 < 2; a2++) {
                if ((unsigned)qy[a2] >= (unsigned)Hs) continue;
                #pragma unroll
                for (int d2 = 0; d2 < 2; d2++) {
                    if ((unsigned)qx[d2] >= (unsigned)Hs) continue;
                    accs += wy[a2] * wx[d2] * sp[qy[a2] * Hs + qx[d2]];
                }
            }
            r3[c] += accs;
        }
    }
    for (int c = 0; c < 3; c++) out[(size_t)(b * 3 + c) * HW + p] = r3[c];
}

// ---------------- constant input tile -> f16 act -----------------------------------
__global__ void const_init_kernel(const float* __restrict__ cw, _Float16* __restrict__ act)
{
    int i = blockIdx.x * 256 + threadIdx.x;
    if (i >= 2 * 512 * 16) return;
    act[i] = (_Float16)cw[i % (512 * 16)];
}

// =================================== host =========================================
struct ConvP { const float *w, *mod_w, *mod_b, *noise_w, *act_b; };
struct RgbP  { const float *w, *mod_w, *mod_b, *bias; };

extern "C" void kernel_launch(void* const* d_in, const int* in_sizes, int n_in,
                              void* d_out, int out_size, void* d_ws, size_t ws_size,
                              hipStream_t stream)
{
    (void)in_sizes; (void)n_in; (void)out_size; (void)ws_size;
    const float** in = (const float**)d_in;

    MlpPtrs mp;
    for (int l = 0; l < 8; l++) { mp.w[l] = in[2 * l]; mp.b[l] = in[2 * l + 1]; }
    const float* constw = in[16];
    ConvP conv1 = { in[17], in[18], in[19], in[20], in[21] };
    RgbP  rgb1  = { in[22], in[23], in[24], in[25] };
    ConvP convs[12];
    for (int i = 0; i < 12; i++) {
        int base = 26 + 5 * i;
        convs[i] = { in[base], in[base + 1], in[base + 2], in[base + 3], in[base + 4] };
    }
    RgbP rgbs[6];
    for (int i = 0; i < 6; i++) {
        int base = 86 + 4 * i;
        rgbs[i] = { in[base], in[base + 1], in[base + 2], in[base + 3] };
    }
    const float* z = in[110];
    const float* noise[13];
    for (int j = 0; j < 13; j++) noise[j] = in[111 + j];

    // ---- workspace carve ----
    char* W = (char*)d_ws;
    size_t off = 0;
    auto carve = [&](size_t bytes) -> void* {
        void* p = W + off;
        off += (bytes + 255) & ~(size_t)255;
        return p;
    };
    float*     wlat  = (float*)carve(2 * 512 * sizeof(float));
    float*     sbuf  = (float*)carve(2 * 512 * sizeof(float));
    float*     srgb  = (float*)carve(2 * 512 * sizeof(float));
    _Float16*  Abuf  = (_Float16*)carve((size_t)2 * 512 * 4608 * sizeof(_Float16));
    _Float16*  act0  = (_Float16*)carve((size_t)2 * 128 * 256 * 256 * sizeof(_Float16));
    _Float16*  act1  = (_Float16*)carve((size_t)2 * 128 * 256 * 256 * sizeof(_Float16));
    float*     gbuf  = (float*)carve((size_t)2 * 512 * 16641 * sizeof(float));
    float*     skip0 = (float*)carve((size_t)2 * 3 * 256 * 256 * sizeof(float));
    float*     skip1 = (float*)carve((size_t)2 * 3 * 256 * 256 * sizeof(float));

    // ---- style network ----
    style_mlp_kernel<<<2, 512, 0, stream>>>(z, mp, wlat);

    auto run_styled = [&](const ConvP& cv, int Cin, int Cout, int Hin, bool up,
                          const float* nz, const _Float16* aIn, _Float16* aOut) {
        style_proj_kernel<<<dim3((Cin + 255) / 256, 2), 256, 0, stream>>>(
            wlat, cv.mod_w, cv.mod_b, sbuf, Cin);
        float scale = 1.f / sqrtf((float)(Cin * 9));
        modulate_kernel<<<dim3(Cout, 2), 256, 0, stream>>>(
            cv.w, sbuf, Abuf, Cin, Cout, up ? 1 : 0, scale);
        int Hout = up ? (2 * Hin + 1) : Hin;
        int Np = Hout * Hout;
        modconv_wmma_kernel<<<dim3((Np + 63) / 64, Cout / 128, 2), 128, 0, stream>>>(
            Abuf, aIn, gbuf, nz, cv.noise_w, cv.act_b, aOut,
            Cin, Cout, Hin, Hin, Hout, Hout, up ? 1 : 0, up ? 0 : 1);
        if (up) {
            int H = 2 * Hin;
            int total = 2 * Cout * H * H;
            blur_act_kernel<<<(total + 255) / 256, 256, 0, stream>>>(
                gbuf, nz, cv.noise_w, cv.act_b, aOut, Cout, H, Hout, total);
        }
    };
    auto run_rgb = [&](const RgbP& rp, const _Float16* a, int Cin, int H, int Hs,
                       const float* skipIn, float* outP) {
        style_proj_kernel<<<dim3((Cin + 255) / 256, 2), 256, 0, stream>>>(
            wlat, rp.mod_w, rp.mod_b, srgb, Cin);
        float scale = 1.f / sqrtf((float)Cin);
        int HW = H * H;
        to_rgb_kernel<<<dim3((HW + 255) / 256, 2), 256, 0, stream>>>(
            a, rp.w, srgb, rp.bias, skipIn, outP, Cin, H, Hs, scale);
    };

    // ---- generator body ----
    const_init_kernel<<<(2 * 512 * 16 + 255) / 256, 256, 0, stream>>>(constw, act0);
    _Float16 *aCur = act0, *aNxt = act1;

    run_styled(conv1, 512, 512, 4, false, noise[0], aCur, aNxt);
    { _Float16* tmp = aCur; aCur = aNxt; aNxt = tmp; }
    run_rgb(rgb1, aCur, 512, 4, 0, nullptr, skip0);

    float *sCur = skip0, *sNxt = skip1;
    int H = 4;
    int cin = 512;
    const int chans[6] = {512, 512, 512, 512, 256, 128};
    for (int i = 0; i < 6; i++) {
        int cout = chans[i];
        run_styled(convs[2 * i], cin, cout, H, true, noise[1 + 2 * i], aCur, aNxt);
        { _Float16* tmp = aCur; aCur = aNxt; aNxt = tmp; }
        H *= 2;
        run_styled(convs[2 * i + 1], cout, cout, H, false, noise[2 + 2 * i], aCur, aNxt);
        { _Float16* tmp = aCur; aCur = aNxt; aNxt = tmp; }
        float* dst = (i == 5) ? (float*)d_out : sNxt;
        run_rgb(rgbs[i], aCur, cout, H, H / 2, sCur, dst);
        if (i != 5) { float* tmp = sCur; sCur = sNxt; sNxt = tmp; }
        cin = cout;
    }
}
